// TrueHigherOrderAttention_39582418600242
// MI455X (gfx1250) — matrix operand order repaired and checked by reading, not compiled, not run on hardware
//
#include <hip/hip_runtime.h>
#include <hip/hip_bf16.h>

// ---------------- problem constants ----------------
#define NB    8
#define NH    16
#define TT    1024
#define HS    64
#define RR    64
#define CC    1024          // NH*HS
#define SCALE 0.125f        // 1/sqrt(64)

typedef __attribute__((ext_vector_type(16))) _Float16 v16h;
typedef __attribute__((ext_vector_type(8)))  _Float16 h8;
typedef __attribute__((ext_vector_type(4)))  _Float16 h4;
typedef __attribute__((ext_vector_type(8)))  float    v8f;

#define WMMA_F16(a, b, c) \
    __builtin_amdgcn_wmma_f32_16x16x32_f16(false, (a), false, (b), (short)0, (c), false, false)

__device__ __forceinline__ v8f zero8() {
    v8f z;
#pragma unroll
    for (int j = 0; j < 8; ++j) z[j] = 0.0f;
    return z;
}

// A fragment: 16x32 f16 tile, row-major source with leading dim ld (multiple of 8).
// ISA A layout per lane(m,half): K = 8*half + 0..7 and 16 + 8*half + 0..7
// -> two contiguous 16-byte loads.
__device__ __forceinline__ v16h load_frag_a(const _Float16* src, int ld, int lane) {
    const int m = lane & 15, half = lane >> 4;
    const h8 lo = *(const h8*)(src + m * ld + 8 * half);
    const h8 hi = *(const h8*)(src + m * ld + 16 + 8 * half);
    return __builtin_shufflevector(lo, hi, 0, 1, 2, 3, 4, 5, 6, 7,
                                   8, 9, 10, 11, 12, 13, 14, 15);
}

// B fragment from an N x K row-major source (i.e. the transpose of the K x N
// operand): lane(n,half) holds K = 8*half + 0..7 and 16 + 8*half + 0..7 -> two 16B loads.
__device__ __forceinline__ v16h load_frag_bt(const _Float16* src, int ld, int lane) {
    const int n = lane & 15, half = lane >> 4;
    const h8 lo = *(const h8*)(src + n * ld + 8 * half);
    const h8 hi = *(const h8*)(src + n * ld + 16 + 8 * half);
    return __builtin_shufflevector(lo, hi, 0, 1, 2, 3, 4, 5, 6, 7,
                                   8, 9, 10, 11, 12, 13, 14, 15);
}

// ---------------- kernel 0: parameter conversion f32 -> f16 (transposed) ----
// segments: WqT0(4096) WqT1(4096) WvT0(4096) coreT(262144) vcoreT(4096) Wp(1048576)
#define CONV_TOTAL (4096 * 4 + 262144 + 1048576)

__global__ void k_convert(const float* __restrict__ qf, const float* __restrict__ qc,
                          const float* __restrict__ vf, const float* __restrict__ vc,
                          const float* __restrict__ wp,
                          _Float16* WqT0, _Float16* WqT1, _Float16* WvT0,
                          _Float16* coreT, _Float16* vcoreT, _Float16* Wp) {
    int i = blockIdx.x * blockDim.x + threadIdx.x;
    if (i >= CONV_TOTAL) return;
    if (i < 4096) { const int r = i >> 6, h = i & 63;           // [r][h] = qf0[h][r]
        WqT0[i] = (_Float16)qf[h * 64 + r]; return; }
    i -= 4096;
    if (i < 4096) { const int r = i >> 6, h = i & 63;
        WqT1[i] = (_Float16)qf[4096 + h * 64 + r]; return; }
    i -= 4096;
    if (i < 4096) { const int r = i >> 6, h = i & 63;
        WvT0[i] = (_Float16)vf[h * 64 + r]; return; }
    i -= 4096;
    if (i < 262144) {                                           // [p][r][q] = qc[p][q][r]
        const int p = i >> 12, rem = i & 4095, r = rem >> 6, q = rem & 63;
        coreT[i] = (_Float16)qc[p * 4096 + q * 64 + r]; return; }
    i -= 262144;
    if (i < 4096) { const int r = i >> 6, p = i & 63;           // [r][p] = vc[p][r]
        vcoreT[i] = (_Float16)vc[p * 64 + r]; return; }
    i -= 4096;
    Wp[i] = (_Float16)wp[i];   // w_proj [j][k] is already the N x K operand for y@W^T
}

// ---------------- kernel 1: projections + Tucker Q, and V^T ----------------
// One wave per 16-token tile of one head. grid = B*NH*(T/16) = 8192.
__global__ __launch_bounds__(32) void k_qv(const float* __restrict__ x,
                                           const _Float16* __restrict__ WqT0,
                                           const _Float16* __restrict__ WqT1,
                                           const _Float16* __restrict__ WvT0,
                                           const _Float16* __restrict__ coreT,
                                           const _Float16* __restrict__ vcoreT,
                                           _Float16* __restrict__ Qh,
                                           _Float16* __restrict__ Vt) {
    __shared__ __align__(16) _Float16 xs[16 * 64];   // x tile, later reused for Q tile
    __shared__ __align__(16) float    q1s[16 * 64];
    __shared__ __align__(16) _Float16 q2s[16 * 64];
    __shared__ __align__(16) _Float16 v1s[16 * 64];

    const int lane = threadIdx.x;
    const int blk  = blockIdx.x;
    const int hd   = blk >> 6;            // b*NH + n
    const int t0   = (blk & 63) << 4;
    const int b    = hd >> 4, hn = hd & 15;
    const int half = lane >> 4, nn = lane & 15;

    // stage x tile [16 x 64] as f16 (float4 loads, 8-byte f16 stores)
    const float* xp = x + ((size_t)(b * TT + t0) * CC) + hn * HS;
#pragma unroll
    for (int i = 0; i < 8; ++i) {
        const int idx = lane + 32 * i;    // 0..255 float4 chunks
        const int m = idx >> 4, c4 = idx & 15;
        const float4 f = *(const float4*)(xp + (size_t)m * CC + c4 * 4);
        h4 h; h[0] = (_Float16)f.x; h[1] = (_Float16)f.y;
              h[2] = (_Float16)f.z; h[3] = (_Float16)f.w;
        *(h4*)&xs[m * 64 + c4 * 4] = h;
    }
    __syncthreads();

    const v16h Ax0 = load_frag_a(xs, 64, lane);
    const v16h Ax1 = load_frag_a(xs + 32, 64, lane);

    // q1 (f32 LDS), q2 (f16 LDS), v1 (f16 LDS): three 16x64x64 GEMMs
#pragma unroll
    for (int t = 0; t < 4; ++t) {
        const int n0 = t * 16;
        v8f f = zero8();
        f = WMMA_F16(Ax0, load_frag_bt(WqT0 + n0 * 64, 64, lane), f);
        f = WMMA_F16(Ax1, load_frag_bt(WqT0 + n0 * 64 + 32, 64, lane), f);
#pragma unroll
        for (int j = 0; j < 8; ++j) q1s[(j + 8 * half) * 64 + n0 + nn] = f[j];

        f = zero8();
        f = WMMA_F16(Ax0, load_frag_bt(WqT1 + n0 * 64, 64, lane), f);
        f = WMMA_F16(Ax1, load_frag_bt(WqT1 + n0 * 64 + 32, 64, lane), f);
#pragma unroll
        for (int j = 0; j < 8; ++j) q2s[(j + 8 * half) * 64 + n0 + nn] = (_Float16)f[j];

        f = zero8();
        f = WMMA_F16(Ax0, load_frag_bt(WvT0 + n0 * 64, 64, lane), f);
        f = WMMA_F16(Ax1, load_frag_bt(WvT0 + n0 * 64 + 32, 64, lane), f);
#pragma unroll
        for (int j = 0; j < 8; ++j) v1s[(j + 8 * half) * 64 + n0 + nn] = (_Float16)f[j];
    }
    __syncthreads();

    // V = v1 @ v_core, stored transposed: Vt[(hd*64 + r)*T + t]
    const v16h Av0 = load_frag_a(v1s, 64, lane);
    const v16h Av1 = load_frag_a(v1s + 32, 64, lane);
#pragma unroll
    for (int t = 0; t < 4; ++t) {
        const int n0 = t * 16;
        v8f f = zero8();
        f = WMMA_F16(Av0, load_frag_bt(vcoreT + n0 * 64, 64, lane), f);
        f = WMMA_F16(Av1, load_frag_bt(vcoreT + n0 * 64 + 32, 64, lane), f);
        h8 hv;
#pragma unroll
        for (int j = 0; j < 8; ++j) hv[j] = (_Float16)f[j];
        // fragment rows j = tokens t0 + 8*half + j (contiguous) -> one 16B store
        *(h8*)(Vt + ((size_t)hd * 64 + n0 + nn) * TT + t0 + 8 * half) = hv;
    }

    // Tucker: Q[t,:] = sum_p q1[t,p] * (q2[t,:] @ core[p])
    const v16h Aq0 = load_frag_a(q2s, 64, lane);
    const v16h Aq1 = load_frag_a(q2s + 32, 64, lane);
    v8f Qa[4];
#pragma unroll
    for (int t = 0; t < 4; ++t) Qa[t] = zero8();

#pragma unroll 1
    for (int p = 0; p < 64; ++p) {
        const _Float16* cp = coreT + p * 4096;       // core[p] as [r][q] (N x K)
        if (p < 63) __builtin_prefetch(coreT + (p + 1) * 4096, 0, 0);
        float s_[8];
#pragma unroll
        for (int j = 0; j < 8; ++j) s_[j] = q1s[(j + 8 * half) * 64 + p];
#pragma unroll
        for (int t = 0; t < 4; ++t) {
            v8f g = zero8();
            g = WMMA_F16(Aq0, load_frag_bt(cp + t * 16 * 64, 64, lane), g);
            g = WMMA_F16(Aq1, load_frag_bt(cp + t * 16 * 64 + 32, 64, lane), g);
#pragma unroll
            for (int j = 0; j < 8; ++j) Qa[t][j] = fmaf(s_[j], g[j], Qa[t][j]);
        }
    }

    // bounce Q tile through LDS (reuse xs) -> coalesced 16B global stores
#pragma unroll
    for (int t = 0; t < 4; ++t)
#pragma unroll
        for (int j = 0; j < 8; ++j)
            xs[(j + 8 * half) * 64 + t * 16 + nn] = (_Float16)Qa[t][j];
    __syncthreads();
    _Float16* Qdst = Qh + ((size_t)hd * TT + t0) * RR;
#pragma unroll
    for (int i = 0; i < 4; ++i) {
        const int chunk = lane + 32 * i;             // 128 h8 chunks
        const int row = chunk >> 3, c8 = chunk & 7;
        *(h8*)&Qdst[row * RR + c8 * 8] = *(const h8*)&xs[row * 64 + c8 * 8];
    }
}

// ---------------- kernel 2: attention (K = Q), softmax, P@V ----------------
// One wave per 16-query tile. grid = B*NH*(T/16) = 8192.
__global__ __launch_bounds__(32) void k_attn(const _Float16* __restrict__ Qh,
                                             const _Float16* __restrict__ Vt,
                                             _Float16* __restrict__ Yh) {
    __shared__ __align__(16) _Float16 pf[16 * 1024]; // scores -> exp'd probs (f16)
    __shared__ float rmax[16];
    __shared__ float rinv[16];

    const int lane = threadIdx.x;
    const int blk  = blockIdx.x;
    const int hd   = blk >> 6;
    const int t0   = (blk & 63) << 4;
    const int half = lane >> 4, nn = lane & 15;

    const _Float16* Qbase = Qh + (size_t)hd * TT * RR;

    const v16h AqA = load_frag_a(Qbase + (size_t)t0 * RR, RR, lane);
    const v16h AqB = load_frag_a(Qbase + (size_t)t0 * RR + 32, RR, lane);

    float rmx[8];
#pragma unroll
    for (int j = 0; j < 8; ++j) rmx[j] = -3.0e38f;

    // S = Q Q^T * scale; key operand is row-major Q = natural N x K source
#pragma unroll 1
    for (int kt = 0; kt < 64; ++kt) {
        v8f s = zero8();
        s = WMMA_F16(AqA, load_frag_bt(Qbase + (size_t)(kt * 16) * RR, RR, lane), s);
        s = WMMA_F16(AqB, load_frag_bt(Qbase + (size_t)(kt * 16) * RR + 32, RR, lane), s);
#pragma unroll
        for (int j = 0; j < 8; ++j) {
            const float v = s[j] * SCALE;
            rmx[j] = fmaxf(rmx[j], v);
            pf[(j + 8 * half) * TT + kt * 16 + nn] = (_Float16)v;
        }
    }
    // reduce row max over the 16 lanes sharing each row
#pragma unroll
    for (int m = 1; m < 16; m <<= 1)
#pragma unroll
        for (int j = 0; j < 8; ++j) rmx[j] = fmaxf(rmx[j], __shfl_xor(rmx[j], m, 16));
    if (nn == 0)
#pragma unroll
        for (int j = 0; j < 8; ++j) rmax[j + 8 * half] = rmx[j];
    __syncthreads();

    // exp + row sum (2 lanes/row, 8-wide vectorized), store exp back into pf
    {
        const int row = lane >> 1, ch = lane & 1;
        const float m = rmax[row];
        float sum = 0.0f;
        for (int c8 = ch * 64; c8 < ch * 64 + 64; ++c8) {
            h8 v = *(const h8*)&pf[row * TT + c8 * 8];
            h8 o;
#pragma unroll
            for (int k = 0; k < 8; ++k) {
                const float e = __expf((float)v[k] - m);
                o[k] = (_Float16)e;
                sum += e;
            }
            *(h8*)&pf[row * TT + c8 * 8] = o;
        }
        sum += __shfl_xor(sum, 1, 32);
        if (ch == 0) rinv[row] = 1.0f / sum;
    }
    __syncthreads();

    // O = P @ V ; V stored transposed [r][t] = natural N x K source
    v8f O[4];
#pragma unroll
    for (int t = 0; t < 4; ++t) O[t] = zero8();

#pragma unroll 1
    for (int ks = 0; ks < 32; ++ks) {
        const v16h Ap = load_frag_a(pf + ks * 32, TT, lane);
        const _Float16* Vb = Vt + (size_t)hd * 64 * TT + (size_t)ks * 32;
#pragma unroll
        for (int t = 0; t < 4; ++t)
            O[t] = WMMA_F16(Ap, load_frag_bt(Vb + (size_t)(t * 16) * TT, TT, lane), O[t]);
    }

    // normalization folded into D scale; bounce tile through LDS for 16B stores
    const int b = hd >> 4, hn = hd & 15;
#pragma unroll
    for (int t = 0; t < 4; ++t)
#pragma unroll
        for (int j = 0; j < 8; ++j)
            pf[(j + 8 * half) * 64 + t * 16 + nn] =
                (_Float16)(O[t][j] * rinv[j + 8 * half]);
    __syncthreads();
    _Float16* Ydst = Yh + ((size_t)b * TT + t0) * CC + hn * HS;
#pragma unroll
    for (int i = 0; i < 4; ++i) {
        const int chunk = lane + 32 * i;
        const int row = chunk >> 3, c8 = chunk & 7;
        *(h8*)&Ydst[(size_t)row * CC + c8 * 8] = *(const h8*)&pf[row * 64 + c8 * 8];
    }
}

// ---------------- kernel 3: output projection out = Y @ W^T ----------------
// One wave per 16x64 output tile. grid = (8192/16)*(1024/64) = 8192.
__global__ __launch_bounds__(32) void k_proj(const _Float16* __restrict__ Yh,
                                             const _Float16* __restrict__ Wp,
                                             float* __restrict__ out) {
    __shared__ __align__(16) float os[16 * 64];

    const int lane = threadIdx.x;
    const int blk  = blockIdx.x;
    const int r0   = (blk >> 4) * 16;     // row tile over B*T = 8192 rows
    const int n0   = (blk & 15) * 64;     // col tile
    const int half = lane >> 4, nn = lane & 15;

    v8f acc[4];
#pragma unroll
    for (int t = 0; t < 4; ++t) acc[t] = zero8();

#pragma unroll 1
    for (int ks = 0; ks < 32; ++ks) {
        const v16h Aa = load_frag_a(Yh + (size_t)r0 * CC + ks * 32, CC, lane);
        const _Float16* Bb = Wp + (size_t)n0 * CC + ks * 32;   // w_proj rows j = N dim
#pragma unroll
        for (int t = 0; t < 4; ++t)
            acc[t] = WMMA_F16(Aa, load_frag_bt(Bb + (size_t)(t * 16) * CC, CC, lane), acc[t]);
    }
#pragma unroll
    for (int t = 0; t < 4; ++t)
#pragma unroll
        for (int j = 0; j < 8; ++j)
            os[(j + 8 * half) * 64 + t * 16 + nn] = acc[t][j];
    __syncthreads();
#pragma unroll
    for (int i = 0; i < 8; ++i) {
        const int idx = lane + 32 * i;    // 256 float4 chunks
        const int row = idx >> 4, c4 = idx & 15;
        *(float4*)&out[(size_t)(r0 + row) * CC + n0 + c4 * 4] =
            *(const float4*)&os[row * 64 + c4 * 4];
    }
}

// ---------------- host-side launch ----------------
extern "C" void kernel_launch(void* const* d_in, const int* in_sizes, int n_in,
                              void* d_out, int out_size, void* d_ws, size_t ws_size,
                              hipStream_t stream) {
    const float* x  = (const float*)d_in[0];   // [B,T,C]
    const float* qf = (const float*)d_in[1];   // [3,64,64]
    const float* qc = (const float*)d_in[2];   // [64,64,64]
    const float* vf = (const float*)d_in[3];   // [2,64,64]
    const float* vc = (const float*)d_in[4];   // [64,64]
    const float* wp = (const float*)d_in[5];   // [C,C]

    // workspace layout (f16 elements)
    _Float16* WqT0   = (_Float16*)d_ws;
    _Float16* WqT1   = WqT0 + 4096;
    _Float16* WvT0   = WqT1 + 4096;
    _Float16* coreT  = WvT0 + 4096;            // 262144
    _Float16* vcoreT = coreT + 262144;         // 4096
    _Float16* Wp     = vcoreT + 4096;          // 1048576
    _Float16* Qh     = Wp + 1048576;           // B*NH*T*R = 8388608  (row-major [t][r])
    _Float16* Vt     = Qh + 8388608;           // 8388608             (transposed [r][t])
    _Float16* Yh     = Vt + 8388608;           // B*T*C  = 8388608

    k_convert<<<(CONV_TOTAL + 255) / 256, 256, 0, stream>>>(qf, qc, vf, vc, wp,
                                                            WqT0, WqT1, WvT0,
                                                            coreT, vcoreT, Wp);
    k_qv  <<<NB * NH * (TT / 16), 32, 0, stream>>>(x, WqT0, WqT1, WvT0, coreT, vcoreT, Qh, Vt);
    k_attn<<<NB * NH * (TT / 16), 32, 0, stream>>>(Qh, Vt, Yh);
    k_proj<<<(NB * TT / 16) * (CC / 64), 32, 0, stream>>>(Yh, Wp, (float*)d_out);
}
